// VectorQuantizer_13322988552765
// MI455X (gfx1250) — compile-verified, matrix operand order checked by reading, not run on hardware
//
#include <hip/hip_runtime.h>
#include <hip/hip_bf16.h>
#include <math.h>

// ---------------------------------------------------------------------------
// VQ-VAE vector quantizer forward for MI455X (gfx1250), wave32 + WMMA.
// Sizes: B=64, D=512, T=768, K=1024  ->  N = B*T = 49152 rows.
// Distances computed with split-fp16 WMMA (hi/lo) for ~fp32-accurate argmin.
// ---------------------------------------------------------------------------

typedef _Float16 half8  __attribute__((ext_vector_type(8)));
typedef _Float16 half16 __attribute__((ext_vector_type(16)));
typedef float    f32x8  __attribute__((ext_vector_type(8)));

#define VQ_B   64
#define VQ_D   512
#define VQ_T   768
#define VQ_K   1024
#define VQ_N   (VQ_B * VQ_T)          // 49152
#define TM     128                    // rows (t's) per block
#define LDS_STRIDE 520                // halves per LDS row (512 + 8 pad -> 1040B)

union H16 { half16 v; half8 h[2]; };

// ---------------------------------------------------------------------------
// Prep: weight [K,512] f32 -> fp16 hi/lo pair + per-code squared norm (f32).
// ---------------------------------------------------------------------------
__global__ __launch_bounds__(256)
void vq_prep(const float* __restrict__ w,
             _Float16* __restrict__ w_hi,
             _Float16* __restrict__ w_lo,
             float* __restrict__ wsq)
{
    __shared__ float red[256];
    const int code = blockIdx.x;
    const int tid  = threadIdx.x;
    float s = 0.f;
    for (int i = tid; i < VQ_D; i += 256) {
        float f = w[code * VQ_D + i];
        _Float16 h = (_Float16)f;
        w_hi[code * VQ_D + i] = h;
        w_lo[code * VQ_D + i] = (_Float16)(f - (float)h);
        s += f * f;
    }
    red[tid] = s;
    __syncthreads();
    for (int off = 128; off > 0; off >>= 1) {
        if (tid < off) red[tid] += red[tid + off];
        __syncthreads();
    }
    if (tid == 0) wsq[code] = red[0];
}

// ---------------------------------------------------------------------------
// Main: stage 128 x-rows (hi+lo fp16) into LDS with transpose, then each of
// 8 waves runs a 16x1024 distance row-block via v_wmma_f32_16x16x32_f16
// (3 WMMAs per tile: hi*hi + hi*lo + lo*hi), tracking per-row argmin.
// ---------------------------------------------------------------------------
__global__ __launch_bounds__(256)
void vq_main(const float* __restrict__ x,        // [B, D, T]
             const _Float16* __restrict__ w_hi,  // [K, D]
             const _Float16* __restrict__ w_lo,  // [K, D]
             const float* __restrict__ wsq,      // [K]
             int* __restrict__ idx_out,          // [N]
             float* __restrict__ counts,         // [K]
             float* __restrict__ loss_accum)     // [1]
{
    extern __shared__ char smem[];
    _Float16* xhi = (_Float16*)smem;                       // 128*520 halves
    _Float16* xlo = xhi + TM * LDS_STRIDE;                 // 128*520 halves
    float*    xsq = (float*)(xlo + TM * LDS_STRIDE);       // 128
    float*    part = xsq + TM;                             // 256

    const int tid = threadIdx.x;
    const int blk = blockIdx.x;
    const int tilesPerB = VQ_T / TM;                       // 6
    const int b  = blk / tilesPerB;
    const int t0 = (blk % tilesPerB) * TM;

    // ---- stage x tile into LDS (transpose [D,T] -> [t,d]), split hi/lo ----
    const int tl   = tid >> 1;        // 0..127  (local t)
    const int halfd = tid & 1;        // which half of D
    const float* src = x + (size_t)b * VQ_D * VQ_T + t0 + tl;
    float ss = 0.f;
    #pragma unroll 4
    for (int i = 0; i < VQ_D / 2; ++i) {
        int d = halfd * (VQ_D / 2) + i;
        float f = src[(size_t)d * VQ_T];
        _Float16 h = (_Float16)f;
        xhi[tl * LDS_STRIDE + d] = h;
        xlo[tl * LDS_STRIDE + d] = (_Float16)(f - (float)h);
        ss += f * f;
    }
    part[tid] = ss;
    __syncthreads();
    if (tid < TM) xsq[tid] = part[2 * tid] + part[2 * tid + 1];
    __syncthreads();

    // ---- WMMA distance sweep ----
    const int wave = tid >> 5;
    const int lane = tid & 31;
    const int lrow = lane & 15;
    const int k0   = (lane >> 4) * 8;          // per-lane K sub-offset (0 or 8)
    const int arow = wave * 16 + lrow;         // local A row this lane feeds

    const _Float16* xhiRow = xhi + arow * LDS_STRIDE;
    const _Float16* xloRow = xlo + arow * LDS_STRIDE;

    float minv[8];
    int   mini[8];
    #pragma unroll
    for (int r = 0; r < 8; ++r) { minv[r] = 3.4e38f; mini[r] = 0; }

    const f32x8 vzero = {0.f, 0.f, 0.f, 0.f, 0.f, 0.f, 0.f, 0.f};

    for (int kg = 0; kg < VQ_K / 64; ++kg) {       // 16 groups of 4 k-tiles
        f32x8 acc[4];
        #pragma unroll
        for (int t = 0; t < 4; ++t) acc[t] = vzero;

        for (int dc = 0; dc < VQ_D / 32; ++dc) {   // 16 d-chunks of 32
            const int dbase = dc * 32 + k0;
            H16 Ahi, Alo;
            Ahi.h[0] = *(const half8*)(xhiRow + dbase);
            Ahi.h[1] = *(const half8*)(xhiRow + dbase + 16);
            Alo.h[0] = *(const half8*)(xloRow + dbase);
            Alo.h[1] = *(const half8*)(xloRow + dbase + 16);

            #pragma unroll
            for (int t = 0; t < 4; ++t) {
                const int code = (kg * 4 + t) * 16 + lrow;
                const _Float16* bh = w_hi + code * VQ_D + dbase;
                const _Float16* bl = w_lo + code * VQ_D + dbase;
                H16 Bhi, Blo;
                Bhi.h[0] = *(const half8*)bh;
                Bhi.h[1] = *(const half8*)(bh + 16);
                Blo.h[0] = *(const half8*)bl;
                Blo.h[1] = *(const half8*)(bl + 16);
                acc[t] = __builtin_amdgcn_wmma_f32_16x16x32_f16(
                    false, Ahi.v, false, Bhi.v, (short)0, acc[t], false, false);
                acc[t] = __builtin_amdgcn_wmma_f32_16x16x32_f16(
                    false, Ahi.v, false, Blo.v, (short)0, acc[t], false, false);
                acc[t] = __builtin_amdgcn_wmma_f32_16x16x32_f16(
                    false, Alo.v, false, Bhi.v, (short)0, acc[t], false, false);
            }
        }

        // fold into running per-row argmin: s(k) = ||w_k||^2 - 2 x.w_k
        #pragma unroll
        for (int t = 0; t < 4; ++t) {
            const int code = (kg * 4 + t) * 16 + lrow;
            const float wq = wsq[code];
            #pragma unroll
            for (int r = 0; r < 8; ++r) {
                float val = wq - 2.0f * acc[t][r];
                bool lt = val < minv[r];
                minv[r] = lt ? val : minv[r];
                mini[r] = lt ? code : mini[r];
            }
        }
    }

    // ---- cross-lane (16-lane segment) lexicographic min reduction ----
    float lsum = 0.f;
    #pragma unroll
    for (int r = 0; r < 8; ++r) {
        float v = minv[r];
        int   i = mini[r];
        #pragma unroll
        for (int m = 1; m < 16; m <<= 1) {
            float vo = __shfl_xor(v, m, 32);
            int   io = __shfl_xor(i, m, 32);
            if (vo < v || (vo == v && io < i)) { v = vo; i = io; }
        }
        if (lrow == 0) {   // lanes 0 (rows r) and 16 (rows 8+r) hold results
            int rloc = wave * 16 + ((lane >> 4) << 3) + r;
            int n = b * VQ_T + t0 + rloc;
            idx_out[n] = i;
            atomicAdd(&counts[i], 1.0f);
            lsum += xsq[rloc] + v;     // ||x||^2 + ||w||^2 - 2 x.w = ||x-w||^2
        }
    }
    if (lrow == 0) atomicAdd(loss_accum, lsum);
}

// ---------------------------------------------------------------------------
// encodings: ones scatter (zeros done via hipMemsetAsync)
// ---------------------------------------------------------------------------
__global__ __launch_bounds__(256)
void vq_scatter(const int* __restrict__ idx, float* __restrict__ enc)
{
    int n = blockIdx.x * blockDim.x + threadIdx.x;
    if (n < VQ_N) enc[(size_t)n * VQ_K + idx[n]] = 1.0f;
}

// ---------------------------------------------------------------------------
// quantized_st output [B, D, T]: out[b,d,t] = weight[idx[b*T+t], d]
// ---------------------------------------------------------------------------
__global__ __launch_bounds__(256)
void vq_gather(const float* __restrict__ w, const int* __restrict__ idx,
               float* __restrict__ outq)
{
    size_t gid = (size_t)blockIdx.x * blockDim.x + threadIdx.x;
    if (gid >= (size_t)VQ_B * VQ_D * VQ_T) return;
    int t = (int)(gid % VQ_T);
    size_t rem = gid / VQ_T;
    int d = (int)(rem % VQ_D);
    int b = (int)(rem / VQ_D);
    int n = b * VQ_T + t;
    outq[gid] = w[(size_t)idx[n] * VQ_D + d];
}

__global__ __launch_bounds__(256)
void vq_idx_to_f(const int* __restrict__ idx, float* __restrict__ o)
{
    int n = blockIdx.x * blockDim.x + threadIdx.x;
    if (n < VQ_N) o[n] = (float)idx[n];
}

// ---------------------------------------------------------------------------
// finalize: loss scalar + perplexity from code histogram
// ---------------------------------------------------------------------------
__global__ __launch_bounds__(1024)
void vq_final(const float* __restrict__ counts,
              const float* __restrict__ loss_accum,
              float* __restrict__ out_loss,
              float* __restrict__ out_perp)
{
    __shared__ float red[1024];
    int k = threadIdx.x;
    float p = counts[k] * (1.0f / (float)VQ_N);
    red[k] = p * logf(p + 1e-10f);
    __syncthreads();
    for (int off = 512; off > 0; off >>= 1) {
        if (k < off) red[k] += red[k + off];
        __syncthreads();
    }
    if (k == 0) {
        *out_perp = expf(-red[0]);
        // loss = q_latent + 0.25 * e_latent = 1.25 * mse  (fwd values equal)
        *out_loss = 1.25f * (*loss_accum) / ((float)VQ_N * (float)VQ_D);
    }
}

// ---------------------------------------------------------------------------
extern "C" void kernel_launch(void* const* d_in, const int* in_sizes, int n_in,
                              void* d_out, int out_size, void* d_ws, size_t ws_size,
                              hipStream_t stream)
{
    (void)in_sizes; (void)n_in; (void)out_size; (void)ws_size;
    const float* x = (const float*)d_in[0];   // [B, D, T]
    const float* w = (const float*)d_in[1];   // [K, D]
    float* out = (float*)d_out;

    // workspace layout (~2.2 MB)
    char* ws = (char*)d_ws;
    _Float16* w_hi  = (_Float16*)(ws + 0);          // 1,048,576 B
    _Float16* w_lo  = (_Float16*)(ws + 1048576);    // 1,048,576 B
    float*    wsq   = (float*)(ws + 2097152);       // 4,096 B
    float*    cnts  = (float*)(ws + 2101248);       // 4,096 B
    float*    lossa = (float*)(ws + 2105344);       // 256 B (only [0] used)
    int*      idx   = (int*)(ws + 2105600);         // 196,608 B

    // output offsets (floats), in reference return order
    const size_t o_q    = 1;
    const size_t o_perp = (size_t)1 + (size_t)VQ_B * VQ_D * VQ_T;      // 25165825
    const size_t o_w    = o_perp + 1;                                   // 25165826
    const size_t o_idx  = o_w + (size_t)VQ_K * VQ_D;                    // 25690114
    const size_t o_enc  = o_idx + (size_t)VQ_N;                         // 25739266

    // zero accumulators + encodings block (memset nodes are graph-capturable)
    hipMemsetAsync(cnts, 0, 4096 + 256, stream);
    hipMemsetAsync(out + o_enc, 0, (size_t)VQ_N * VQ_K * sizeof(float), stream);

    // 1) weight -> fp16 hi/lo + ||w||^2
    vq_prep<<<VQ_K, 256, 0, stream>>>(w, w_hi, w_lo, wsq);

    // 2) main WMMA distance + argmin pass (268 KB dynamic LDS; WGP has 320 KB)
    const size_t smem_bytes = (size_t)2 * TM * LDS_STRIDE * sizeof(_Float16)
                            + (TM + 256) * sizeof(float);               // 267,776
    hipFuncSetAttribute((const void*)vq_main,
                        hipFuncAttributeMaxDynamicSharedMemorySize,
                        (int)smem_bytes);
    vq_main<<<VQ_B * (VQ_T / TM), 256, smem_bytes, stream>>>(
        x, w_hi, w_lo, wsq, idx, cnts, lossa);

    // 3) one-hot encodings ones
    vq_scatter<<<(VQ_N + 255) / 256, 256, 0, stream>>>(idx, out + o_enc);

    // 4) quantized_st gather back to [B, D, T]
    vq_gather<<<(int)(((size_t)VQ_B * VQ_D * VQ_T + 255) / 256), 256, 0, stream>>>(
        w, idx, out + o_q);

    // 5) weight passthrough output
    hipMemcpyAsync(out + o_w, w, (size_t)VQ_K * VQ_D * sizeof(float),
                   hipMemcpyDeviceToDevice, stream);

    // 6) indices as float
    vq_idx_to_f<<<(VQ_N + 255) / 256, 256, 0, stream>>>(idx, out + o_idx);

    // 7) loss + perplexity scalars
    vq_final<<<1, 1024, 0, stream>>>(cnts, lossa, out + 0, out + o_perp);
}
